// EncoderLayer_23725399343590
// MI455X (gfx1250) — matrix operand order repaired and checked by reading, not compiled, not run on hardware
//
#include <hip/hip_runtime.h>
#include <hip/hip_bf16.h>
#include <stdint.h>

typedef _Float16 v16h __attribute__((ext_vector_type(16)));
typedef _Float16 v8h  __attribute__((ext_vector_type(8)));
typedef float    v8f  __attribute__((ext_vector_type(8)));

#define RTOT  8192   // B*N rows
#define EDIM  256
#define IDIM  768
#define NSEQ  2048
#define BATCH 4
#define NHEAD 8
#define DHEAD 32
#define NEGV  -1.0e9f

// ---- CDNA5 WMMA 16-bit fragment maps (ISA 7.12.2) ----
// A (16x32): lane l -> row l&15, K chunks {c0..c0+7},{c0+16..c0+23}, c0=(l<16?0:8)
// B (32x16): lane l -> col l&15, same K chunks as A; this code loads A and B K-contiguous kb..kb+15, kb=(l<16?0:16)
// C (16x16): lane l, vgpr i -> row i+(l>=16?8:0), col l&15

__device__ __forceinline__ v16h cat8(v8h lo, v8h hi) {
  return __builtin_shufflevector(lo, hi, 0, 1, 2, 3, 4, 5, 6, 7, 8, 9, 10, 11, 12, 13, 14, 15);
}

// async global->LDS 16B copy (GLOBAL_LOAD_ASYNC_TO_LDS_B128, ASYNCcnt-tracked)
__device__ __forceinline__ void async_ld_b128(uint32_t lds_off, const _Float16* g) {
  asm volatile("global_load_async_to_lds_b128 %0, %1, off"
               :: "v"(lds_off), "v"((uint64_t)(uintptr_t)g) : "memory");
}
__device__ __forceinline__ void wait_async0() {
  asm volatile("s_wait_asynccnt 0x0" ::: "memory");
}

__global__ void cast_f32_f16(const float* __restrict__ in, _Float16* __restrict__ out, int n) {
  int i = blockIdx.x * blockDim.x + threadIdx.x;
  if (i < n) out[i] = (_Float16)in[i];
}

// in[R][C] f32 -> out[C][R] f16  (N-major weight copies for contiguous B fragments)
__global__ void cast_transpose_f32_f16(const float* __restrict__ in, _Float16* __restrict__ out,
                                       int R, int C) {
  int i = blockIdx.x * blockDim.x + threadIdx.x;
  if (i < R * C) {
    int r = i / C, c = i % C;
    out[(size_t)c * R + r] = (_Float16)in[i];
  }
}

// Vt[(b*256 + c)*2048 + n] = Vh[(b*2048 + n)*256 + c]   (c = h*32+d)
__global__ void transpose_v(const _Float16* __restrict__ Vh, _Float16* __restrict__ Vt) {
  int i = blockIdx.x * blockDim.x + threadIdx.x;   // over BATCH*256*2048
  int n = i & (NSEQ - 1);
  int bc = i >> 11;
  int b = bc >> 8, c = bc & 255;
  Vt[i] = Vh[((size_t)b * NSEQ + n) * EDIM + c];
}

// C[M,N] = A[M,K] x B[K,N]; A row-major f16, Bt = B transposed (N-major) f16; f32 accum.
// Block: 256 thr = 8 waves (4x2). Block tile 128x128; wave tile 32x64 = 8 WMMA tiles.
// A panel double-buffered in LDS via async b128 copies (overlapped with WMMA);
// B fragments straight from L2 (weights are tiny), batched ahead of the WMMA burst.
__global__ void wmma_gemm_f16(const _Float16* __restrict__ A, const _Float16* __restrict__ Bt,
                              float* __restrict__ Cf, _Float16* __restrict__ Ch,
                              const float* __restrict__ bias,
                              int M, int N, int K, int out_half) {
  __shared__ _Float16 As[2][128][40];
  const int tid = threadIdx.x, lane = tid & 31, wave = tid >> 5;
  const int wm = wave & 3, wn = wave >> 2;
  const int bm0 = blockIdx.x * 128, bn0 = blockIdx.y * 128;

  v8f acc0[4] = {};   // rows wm*32 .. +16
  v8f acc1[4] = {};   // rows wm*32+16 .. +32

  // async staging: thread t -> row t>>1 (0..127), half-row chunk (t&1)*16 (two b128s)
  const int sr = tid >> 1, sc = (tid & 1) * 16;
  const uint32_t ldsd[2][2] = {
    { (uint32_t)(uintptr_t)&As[0][sr][sc], (uint32_t)(uintptr_t)&As[0][sr][sc + 8] },
    { (uint32_t)(uintptr_t)&As[1][sr][sc], (uint32_t)(uintptr_t)&As[1][sr][sc + 8] } };
  const _Float16* Asrc = A + (size_t)(bm0 + sr) * K + sc;

  const int c0a = (lane < 16) ? 0 : 8;
  const int kb  = (lane < 16) ? 0 : 16;
  const int bnl = bn0 + wn * 64 + (lane & 15);
  const int ar0 = wm * 32 + (lane & 15);

  const int nk = K / 32;
  // prologue: stage panel 0
  async_ld_b128(ldsd[0][0], Asrc);
  async_ld_b128(ldsd[0][1], Asrc + 8);
  wait_async0();
  __syncthreads();

  for (int i = 0; i < nk; ++i) {
    if (i + 1 < nk) {          // prefetch next panel into the other buffer (overlaps WMMAs)
      const _Float16* s = Asrc + (i + 1) * 32;
      async_ld_b128(ldsd[(i + 1) & 1][0], s);
      async_ld_b128(ldsd[(i + 1) & 1][1], s + 8);
    }
    const _Float16(*Ab)[40] = As[i & 1];
    const int k0 = i * 32;

    v16h bf[4];
#pragma unroll
    for (int t = 0; t < 4; ++t) {
      const _Float16* bp = Bt + (size_t)(bnl + t * 16) * K + k0 + kb;
      bf[t] = cat8(*(const v8h*)bp, *(const v8h*)(bp + 8));
    }
    v16h af0 = cat8(*(const v8h*)&Ab[ar0][kb],      *(const v8h*)&Ab[ar0][kb + 8]);
    v16h af1 = cat8(*(const v8h*)&Ab[ar0 + 16][kb], *(const v8h*)&Ab[ar0 + 16][kb + 8]);
#pragma unroll
    for (int t = 0; t < 4; ++t) {
      acc0[t] = __builtin_amdgcn_wmma_f32_16x16x32_f16(false, af0, false, bf[t], (short)0, acc0[t], false, false);
      acc1[t] = __builtin_amdgcn_wmma_f32_16x16x32_f16(false, af1, false, bf[t], (short)0, acc1[t], false, false);
    }
    wait_async0();
    __syncthreads();
  }

  const int rb0 = bm0 + wm * 32 + ((lane >= 16) ? 8 : 0);
#pragma unroll
  for (int t = 0; t < 4; ++t) {
    const int c = bn0 + wn * 64 + t * 16 + (lane & 15);
    const float bv = bias ? bias[c] : 0.0f;
#pragma unroll
    for (int i = 0; i < 8; ++i) {
      float v0 = acc0[t][i] + bv;
      float v1 = acc1[t][i] + bv;
      size_t o0 = (size_t)(rb0 + i) * N + c;
      size_t o1 = (size_t)(rb0 + 16 + i) * N + c;
      if (out_half) { Ch[o0] = (_Float16)v0; Ch[o1] = (_Float16)v1; }
      else          { Cf[o0] = v0;           Cf[o1] = v1; }
    }
  }
}

// Fused top-k attention. One block per (16-query tile, head, batch); 256 threads (8 waves).
// Dynamic LDS: S[16][2048] f32 (128KB, reused for cross-wave reduce) + W[16][2048] f16 (64KB).
__global__ void attn_topk(const _Float16* __restrict__ Qh, const _Float16* __restrict__ Kh,
                          const _Float16* __restrict__ Vt, const int* __restrict__ amask,
                          _Float16* __restrict__ Oh) {
  extern __shared__ char smem[];
  float*    S       = (float*)smem;                                   // [16][NSEQ]
  _Float16* W       = (_Float16*)(smem + 16 * NSEQ * sizeof(float));  // [16][NSEQ]
  float*    partial = (float*)smem;                                   // reuse: [8][16][32]

  const int lane = threadIdx.x & 31;
  const int wave = threadIdx.x >> 5;
  const int qt = blockIdx.x, h = blockIdx.y, b = blockIdx.z;
  const int n0 = qt * 16;
  const size_t bbase = (size_t)b * NSEQ;
  const float scale = 0.17677669529663687f; // 1/sqrt(32)
  const int c0a = (lane < 16) ? 0 : 8;
  const int kb  = (lane < 16) ? 0 : 16;

  // ---- phase 1: S = mask(Q K^T / sqrt(D)); K^T is already N-major in Kh -> b128 loads
  v16h qf;
  {
    const _Float16* qb = Qh + (bbase + n0 + (lane & 15)) * EDIM + h * DHEAD;
    qf = cat8(*(const v8h*)(qb + kb), *(const v8h*)(qb + kb + 8));
  }
  for (int kt = wave; kt < NSEQ / 16; kt += 8) {
    const int m0 = kt * 16;
    const _Float16* kp = Kh + (bbase + m0 + (lane & 15)) * EDIM + h * DHEAD + kb;
    v16h kf = cat8(*(const v8h*)kp, *(const v8h*)(kp + 8));
    v8f c = {};
    c = __builtin_amdgcn_wmma_f32_16x16x32_f16(false, qf, false, kf, (short)0, c, false, false);
    const int col  = lane & 15;
    const int mg   = m0 + col;
    const int keep = amask[bbase + mg];
#pragma unroll
    for (int i = 0; i < 8; ++i) {
      int r = i + ((lane >= 16) ? 8 : 0);
      S[r * NSEQ + mg] = keep ? (c[i] * scale) : NEGV;
    }
  }
  __syncthreads();

  // ---- phase 2: stage row into 64 VGPRs/lane; bisection top-1024 + softmax -> W (f16)
#pragma unroll 1
  for (int rr = 0; rr < 2; ++rr) {
    const int r = wave * 2 + rr;
    const float* row = S + r * NSEQ;
    float rv[64];
#pragma unroll
    for (int j = 0; j < 64; ++j) rv[j] = row[lane + 32 * j];
    float mx = NEGV;
#pragma unroll
    for (int j = 0; j < 64; ++j) mx = fmaxf(mx, rv[j]);
#pragma unroll
    for (int off = 16; off; off >>= 1) mx = fmaxf(mx, __shfl_xor(mx, off));
    float lo = mx - 64.0f, hi = mx;    // below mx-64 exp() underflows anyway
    for (int it = 0; it < 20; ++it) {
      float mid = 0.5f * (lo + hi);
      int cnt = 0;
#pragma unroll
      for (int j = 0; j < 64; ++j) cnt += (rv[j] >= mid) ? 1 : 0;
#pragma unroll
      for (int off = 16; off; off >>= 1) cnt += __shfl_xor(cnt, off);
      if (cnt >= NSEQ / 2) lo = mid; else hi = mid;
    }
    const float thr = lo;
    float ssum = 0.0f;
#pragma unroll
    for (int j = 0; j < 64; ++j) {
      float e = (rv[j] >= thr) ? __expf(rv[j] - mx) : 0.0f;
      rv[j] = e;
      ssum += e;
    }
#pragma unroll
    for (int off = 16; off; off >>= 1) ssum += __shfl_xor(ssum, off);
    const float inv = 1.0f / ssum;
#pragma unroll
    for (int j = 0; j < 64; ++j) W[r * NSEQ + lane + 32 * j] = (_Float16)(rv[j] * inv);
  }
  __syncthreads();

  // ---- phase 3: O[16,32] = W[16,2048] @ V[2048,32]; V read N-major from Vt -> b128 loads
  v8f o0 = {}; v8f o1 = {};
  const int kb0 = wave * (NSEQ / 8);
  const _Float16* v0base = Vt + (size_t)(b * 256 + h * DHEAD + (lane & 15)) * NSEQ;
  const _Float16* v1base = v0base + (size_t)16 * NSEQ;
  const _Float16* wrow   = W + (lane & 15) * NSEQ;
  for (int kk = 0; kk < NSEQ / 8; kk += 32) {
    const int kpos = kb0 + kk;
    v16h wf  = cat8(*(const v8h*)(wrow + kpos + kb), *(const v8h*)(wrow + kpos + kb + 8));
    v16h vf0 = cat8(*(const v8h*)(v0base + kpos + kb), *(const v8h*)(v0base + kpos + kb + 8));
    v16h vf1 = cat8(*(const v8h*)(v1base + kpos + kb), *(const v8h*)(v1base + kpos + kb + 8));
    o0 = __builtin_amdgcn_wmma_f32_16x16x32_f16(false, wf, false, vf0, (short)0, o0, false, false);
    o1 = __builtin_amdgcn_wmma_f32_16x16x32_f16(false, wf, false, vf1, (short)0, o1, false, false);
  }
  float* pw = partial + wave * (16 * 32);
#pragma unroll
  for (int i = 0; i < 8; ++i) {
    int r = i + ((lane >= 16) ? 8 : 0);
    int c = lane & 15;
    pw[r * 32 + c]      = o0[i];
    pw[r * 32 + 16 + c] = o1[i];
  }
  __syncthreads();
  for (int idx = threadIdx.x; idx < 16 * 32; idx += blockDim.x) {
    int r = idx >> 5, c = idx & 31;
    float acc = 0.0f;
#pragma unroll
    for (int w = 0; w < 8; ++w) acc += partial[w * 512 + r * 32 + c];
    Oh[(bbase + n0 + r) * EDIM + h * DHEAD + c] = (_Float16)acc;
  }
}

// out = rmsnorm(a + b) * w ; optionally also emit f16 copy. One 256-thread block per row.
__global__ void add_rmsnorm(const float* __restrict__ a, const float* __restrict__ bsrc,
                            const float* __restrict__ w, float* __restrict__ of,
                            _Float16* __restrict__ oh) {
  const int row = blockIdx.x, t = threadIdx.x;
  __shared__ float red[256];
  const size_t idx = (size_t)row * EDIM + t;
  float v = a[idx] + bsrc[idx];
  red[t] = v * v;
  __syncthreads();
  for (int off = 128; off; off >>= 1) { if (t < off) red[t] += red[t + off]; __syncthreads(); }
  const float inv = rsqrtf(red[0] * (1.0f / EDIM) + 1e-6f);
  const float o = v * inv * w[t];
  if (of) of[idx] = o;
  if (oh) oh[idx] = (_Float16)o;
}

__global__ void silu_mul(const _Float16* __restrict__ g1, const _Float16* __restrict__ g2,
                         _Float16* __restrict__ out, int n) {
  int i = blockIdx.x * blockDim.x + threadIdx.x;
  if (i < n) {
    float a = (float)g1[i], b = (float)g2[i];
    out[i] = (_Float16)((a / (1.0f + __expf(-a))) * b);
  }
}

extern "C" void kernel_launch(void* const* d_in, const int* in_sizes, int n_in,
                              void* d_out, int out_size, void* d_ws, size_t ws_size,
                              hipStream_t stream) {
  (void)in_sizes; (void)n_in; (void)out_size; (void)ws_size;
  const float* x   = (const float*)d_in[0];
  const int*   am  = (const int*)d_in[1];
  const float* Wq  = (const float*)d_in[2];
  const float* Wk  = (const float*)d_in[3];
  const float* Wv  = (const float*)d_in[4];
  const float* Wo  = (const float*)d_in[5];
  const float* bo  = (const float*)d_in[6];
  const float* n1w = (const float*)d_in[7];
  const float* l1  = (const float*)d_in[8];
  const float* l2  = (const float*)d_in[9];
  const float* l3  = (const float*)d_in[10];
  const float* n2w = (const float*)d_in[11];
  float* out = (float*)d_out;

  char* ws = (char*)d_ws;
  size_t off = 0;
  auto alloc = [&](size_t bytes) {
    char* p = ws + off;
    off = (off + bytes + 255) & ~(size_t)255;
    return p;
  };
  _Float16* xh   = (_Float16*)alloc((size_t)RTOT * EDIM * 2);
  _Float16* wqt  = (_Float16*)alloc((size_t)EDIM * EDIM * 2);   // [N][K]
  _Float16* wkt  = (_Float16*)alloc((size_t)EDIM * EDIM * 2);
  _Float16* wvt  = (_Float16*)alloc((size_t)EDIM * EDIM * 2);
  _Float16* wot  = (_Float16*)alloc((size_t)EDIM * EDIM * 2);
  _Float16* l1t  = (_Float16*)alloc((size_t)EDIM * IDIM * 2);   // [768][256]
  _Float16* l2t  = (_Float16*)alloc((size_t)EDIM * IDIM * 2);
  _Float16* l3t  = (_Float16*)alloc((size_t)IDIM * EDIM * 2);   // [256][768]
  _Float16* Qh   = (_Float16*)alloc((size_t)RTOT * EDIM * 2);
  _Float16* Kh   = (_Float16*)alloc((size_t)RTOT * EDIM * 2);
  _Float16* Vh   = (_Float16*)alloc((size_t)RTOT * EDIM * 2);
  _Float16* Vt   = (_Float16*)alloc((size_t)RTOT * EDIM * 2);
  _Float16* atth = (_Float16*)alloc((size_t)RTOT * EDIM * 2);
  float*    mha  = (float*)   alloc((size_t)RTOT * EDIM * 4);
  float*    hf   = (float*)   alloc((size_t)RTOT * EDIM * 4);
  _Float16* hh   = (_Float16*)alloc((size_t)RTOT * EDIM * 2);
  _Float16* g1h  = (_Float16*)alloc((size_t)RTOT * IDIM * 2);
  _Float16* g2h  = (_Float16*)alloc((size_t)RTOT * IDIM * 2);
  _Float16* ghh  = (_Float16*)alloc((size_t)RTOT * IDIM * 2);
  float*    ff   = (float*)   alloc((size_t)RTOT * EDIM * 4);

  // casts (activations plain, weights transposed to N-major)
  cast_f32_f16<<<(RTOT * EDIM + 255) / 256, 256, 0, stream>>>(x, xh, RTOT * EDIM);
  cast_transpose_f32_f16<<<(EDIM * EDIM + 255) / 256, 256, 0, stream>>>(Wq, wqt, EDIM, EDIM);
  cast_transpose_f32_f16<<<(EDIM * EDIM + 255) / 256, 256, 0, stream>>>(Wk, wkt, EDIM, EDIM);
  cast_transpose_f32_f16<<<(EDIM * EDIM + 255) / 256, 256, 0, stream>>>(Wv, wvt, EDIM, EDIM);
  cast_transpose_f32_f16<<<(EDIM * EDIM + 255) / 256, 256, 0, stream>>>(Wo, wot, EDIM, EDIM);
  cast_transpose_f32_f16<<<(EDIM * IDIM + 255) / 256, 256, 0, stream>>>(l1, l1t, EDIM, IDIM);
  cast_transpose_f32_f16<<<(EDIM * IDIM + 255) / 256, 256, 0, stream>>>(l2, l2t, EDIM, IDIM);
  cast_transpose_f32_f16<<<(IDIM * EDIM + 255) / 256, 256, 0, stream>>>(l3, l3t, IDIM, EDIM);

  // QKV projections -> f16, head-contiguous [B*N, H*D]
  dim3 gE(RTOT / 128, EDIM / 128);
  wmma_gemm_f16<<<gE, 256, 0, stream>>>(xh, wqt, nullptr, Qh, nullptr, RTOT, EDIM, EDIM, 1);
  wmma_gemm_f16<<<gE, 256, 0, stream>>>(xh, wkt, nullptr, Kh, nullptr, RTOT, EDIM, EDIM, 1);
  wmma_gemm_f16<<<gE, 256, 0, stream>>>(xh, wvt, nullptr, Vh, nullptr, RTOT, EDIM, EDIM, 1);
  transpose_v<<<(BATCH * EDIM * NSEQ + 255) / 256, 256, 0, stream>>>(Vh, Vt);

  // fused score / top-k / softmax / value einsum (192KB dynamic LDS per WG)
  size_t attn_lds = (size_t)16 * NSEQ * sizeof(float) + (size_t)16 * NSEQ * sizeof(_Float16);
  attn_topk<<<dim3(NSEQ / 16, NHEAD, BATCH), 256, attn_lds, stream>>>(Qh, Kh, Vt, am, atth);

  // output projection + bias -> f32
  wmma_gemm_f16<<<gE, 256, 0, stream>>>(atth, wot, mha, nullptr, bo, RTOT, EDIM, EDIM, 0);

  // h = rmsnorm(x + mha); keep f32 + f16 views
  add_rmsnorm<<<RTOT, 256, 0, stream>>>(x, mha, n1w, hf, hh);

  // FFN
  dim3 gI(RTOT / 128, IDIM / 128);
  wmma_gemm_f16<<<gI, 256, 0, stream>>>(hh, l1t, nullptr, g1h, nullptr, RTOT, IDIM, EDIM, 1);
  wmma_gemm_f16<<<gI, 256, 0, stream>>>(hh, l2t, nullptr, g2h, nullptr, RTOT, IDIM, EDIM, 1);
  silu_mul<<<(RTOT * IDIM + 255) / 256, 256, 0, stream>>>(g1h, g2h, ghh, RTOT * IDIM);
  wmma_gemm_f16<<<gE, 256, 0, stream>>>(ghh, l3t, ff, nullptr, nullptr, RTOT, EDIM, IDIM, 0);

  // out = rmsnorm(h + ff)
  add_rmsnorm<<<RTOT, 256, 0, stream>>>(hf, ff, n2w, out, nullptr);
}
